// _EnsembleAttentionBlock_45921790329499
// MI455X (gfx1250) — matrix operand order repaired and checked by reading, not compiled, not run on hardware
//
#include <hip/hip_runtime.h>
#include <hip/hip_bf16.h>

// Problem constants
#define B_    64
#define C_    768
#define M_    512
#define NH_   8
#define DH_   96
#define OQKV  2304   // 2*H + C
#define EPS_  1e-5f

typedef __attribute__((ext_vector_type(16))) __bf16  v16bf;
typedef __attribute__((ext_vector_type(8)))  float    v8f;
typedef __attribute__((ext_vector_type(4)))  unsigned v4u;
typedef __attribute__((ext_vector_type(8)))  unsigned v8u;

static __device__ __forceinline__ unsigned short f2bf(float f) {
    unsigned u = __builtin_bit_cast(unsigned, f);
    unsigned r = 0x7FFFu + ((u >> 16) & 1u);       // round-to-nearest-even
    return (unsigned short)((u + r) >> 16);
}
static __device__ __forceinline__ unsigned pk_bf16(float a, float b) {
    return (unsigned)f2bf(a) | ((unsigned)f2bf(b) << 16);
}
static __device__ __forceinline__ v16bf frag_from(v4u lo, v4u hi) {
    v8u u;
    u[0]=lo[0]; u[1]=lo[1]; u[2]=lo[2]; u[3]=lo[3];
    u[4]=hi[0]; u[5]=hi[1]; u[6]=hi[2]; u[7]=hi[3];
    return __builtin_bit_cast(v16bf, u);
}
// A fragment: 16(M) x 32(K) bf16 from row-major [m][k] storage (32 k contiguous = 64B rows)
// lane<16 : K0..15 @ bytes[0,32) ; lane>=16: K16..31 @ [32,64)
static __device__ __forceinline__ v16bf load_a(const char* base, int rowBytes, int lane) {
    int row = lane & 15, half = lane >> 4;
    const char* p = base + (size_t)row * rowBytes + half * 32;
    return frag_from(*(const v4u*)p, *(const v4u*)(p + 16));
}
// B fragment: 32(K) x 16(N) bf16 from [n][k] storage (32 k contiguous)
// lane<16 : K0..15 @ bytes[0,32) ; lane>=16: K16..31 @ [32,64)
static __device__ __forceinline__ v16bf load_b(const char* base, int rowBytes, int lane) {
    int col = lane & 15, half = lane >> 4;
    const char* p = base + (size_t)col * rowBytes + half * 32;
    return frag_from(*(const v4u*)p, *(const v4u*)(p + 16));
}
static __device__ __forceinline__ v8f wmma_bf16(v16bf a, v16bf b, v8f c) {
    return __builtin_amdgcn_wmma_f32_16x16x32_bf16(false, a, false, b, (short)0, c, false, false);
}

// ---------------- Kernel 1: weights fp32 -> bf16 ----------------
__global__ void k_cvt_weights(const float* __restrict__ wq, const float* __restrict__ wp,
                              unsigned short* __restrict__ wqb, unsigned short* __restrict__ wpb) {
    const int n1 = OQKV * C_, n2 = C_ * C_;
    for (int i = blockIdx.x * 256 + threadIdx.x; i < n1 + n2; i += gridDim.x * 256) {
        if (i < n1) wqb[i] = f2bf(wq[i]);
        else        wpb[i - n1] = f2bf(wp[i - n1]);
    }
}

// ---------------- Kernel 2: x (B,C,M) fp32 -> xT (B,M,C) bf16, tiled transpose ----------------
__global__ void k_transpose_x(const float* __restrict__ x, unsigned short* __restrict__ xT) {
    __shared__ unsigned short tile[32][33];
    int b = blockIdx.z, c0 = blockIdx.y * 32, m0 = blockIdx.x * 32;
    #pragma unroll
    for (int k = 0; k < 4; ++k) {
        int c = c0 + threadIdx.y + k * 8;
        tile[threadIdx.y + k * 8][threadIdx.x] =
            f2bf(x[((size_t)b * C_ + c) * M_ + m0 + threadIdx.x]);
    }
    __syncthreads();
    #pragma unroll
    for (int k = 0; k < 4; ++k) {
        int m = m0 + threadIdx.y + k * 8;
        xT[((size_t)b * M_ + m) * C_ + c0 + threadIdx.x] = tile[threadIdx.x][threadIdx.y + k * 8];
    }
}

// ---------------- Kernel 3: qkv GEMM (WMMA) + scatter to qT/kT/v ----------------
// qkv[o][m] = sum_c w_qkv[o][c] * x[c][m]; A = w_qkv [o][c], B = xT [m][c]
// Wave: one 16-o tile x 64 m (4 subtiles). Scatter: qT/kT as [bh][m][d], v as [bh][d][m].
__global__ void k_qkv_gemm(const char* __restrict__ wqb, const char* __restrict__ xT,
                           unsigned short* __restrict__ qT, unsigned short* __restrict__ kT,
                           unsigned short* __restrict__ vm) {
    int lane = threadIdx.x & 31;
    int wid  = blockIdx.x * 8 + (threadIdx.x >> 5);
    int ms = wid & 7;              // m strip (64)
    int ot = (wid >> 3) % 144;     // o tile (16)
    int b  = wid / (144 * 8);
    int o0 = ot * 16, mbase = ms * 64;

    const char* Abase = wqb + (size_t)o0 * (C_ * 2);
    const char* Bbase = xT + ((size_t)b * M_ + mbase) * (C_ * 2);
    v8f acc[4];
    #pragma unroll
    for (int s = 0; s < 4; ++s) acc[s] = (v8f)(0.0f);

    for (int ck = 0; ck < C_ / 32; ++ck) {
        v16bf af = load_a(Abase + ck * 64, C_ * 2, lane);
        #pragma unroll
        for (int s = 0; s < 4; ++s) {
            v16bf bf = load_b(Bbase + (size_t)s * 16 * (C_ * 2) + ck * 64, C_ * 2, lane);
            acc[s] = wmma_bf16(af, bf, acc[s]);
        }
    }
    int lm = lane & 15, oofs = (lane >> 4) * 8;
    #pragma unroll
    for (int s = 0; s < 4; ++s) {
        int m = mbase + s * 16 + lm;
        #pragma unroll
        for (int r = 0; r < 8; ++r) {
            int o = o0 + oofs + r;
            int h = o / 288, rr = o - h * 288;
            unsigned short val = f2bf(acc[s][r]);
            size_t bh = (size_t)b * NH_ + h;
            if (rr < DH_)             qT[(bh * M_ + m) * DH_ + rr]          = val;
            else if (rr < 2 * DH_)    kT[(bh * M_ + m) * DH_ + (rr - DH_)]  = val;
            else                      vm[(bh * DH_ + (rr - 2 * DH_)) * M_ + m] = val;
        }
    }
}

// ---------------- Kernel 4: fused attention (flash-style online softmax, all WMMA) ----------------
// S[i][j] = sum_d kT[i][d]*qT[j][d] / sqrt(768); softmax over i; O[d][j] = sum_i v[d][i]*P[i][j]
// Wave handles one (b,h) and 16 queries j. Writes aT[b][j][h*96+d] bf16.
__global__ void k_attention(const char* __restrict__ qT, const char* __restrict__ kT,
                            const char* __restrict__ vm, unsigned short* __restrict__ aT) {
    int lane = threadIdx.x & 31;
    int wid  = blockIdx.x * 8 + (threadIdx.x >> 5);
    int jt = wid & 31;                 // 32 j-tiles of 16
    int h  = (wid >> 5) & 7;
    int b  = wid >> 8;
    size_t bh = (size_t)b * NH_ + h;
    const char* Kb = kT + bh * M_ * DH_ * 2;   // rows i, 192B
    const char* Qb = qT + bh * M_ * DH_ * 2;   // rows j, 192B
    const char* Vb = vm + bh * DH_ * M_ * 2;   // rows d, 1024B
    int j0 = jt * 16;

    // Q B-fragments are invariant over the i loop
    v16bf qb[3];
    #pragma unroll
    for (int dc = 0; dc < 3; ++dc)
        qb[dc] = load_b(Qb + (size_t)j0 * (DH_ * 2) + dc * 64, DH_ * 2, lane);

    v8f O[6];
    #pragma unroll
    for (int t = 0; t < 6; ++t) O[t] = (v8f)(0.0f);
    float mrun = -3.0e38f, lrun = 0.0f;
    const float sc = 0.03608439182435161f;     // 1/sqrt(768)
    bool hi = lane >= 16;

    for (int u = 0; u < 16; ++u) {             // 32 keys per iteration
        int i0 = u * 32;
        v8f S0 = (v8f)(0.0f), S1 = (v8f)(0.0f);
        #pragma unroll
        for (int dc = 0; dc < 3; ++dc) {
            v16bf ka0 = load_a(Kb + (size_t)i0 * (DH_ * 2) + dc * 64, DH_ * 2, lane);
            v16bf ka1 = load_a(Kb + (size_t)(i0 + 16) * (DH_ * 2) + dc * 64, DH_ * 2, lane);
            S0 = wmma_bf16(ka0, qb[dc], S0);
            S1 = wmma_bf16(ka1, qb[dc], S1);
        }
        float p0[8], p1[8], bm = -3.0e38f;
        #pragma unroll
        for (int r = 0; r < 8; ++r) {
            p0[r] = S0[r] * sc; bm = fmaxf(bm, p0[r]);
            p1[r] = S1[r] * sc; bm = fmaxf(bm, p1[r]);
        }
        bm = fmaxf(bm, __shfl_xor(bm, 16, 32));
        float mnew  = fmaxf(mrun, bm);
        float alpha = __expf(mrun - mnew);
        float psum  = 0.0f;
        #pragma unroll
        for (int r = 0; r < 8; ++r) {
            p0[r] = __expf(p0[r] - mnew);
            p1[r] = __expf(p1[r] - mnew);
            psum += p0[r] + p1[r];
        }
        psum += __shfl_xor(psum, 16, 32);
        lrun = lrun * alpha + psum;
        mrun = mnew;
        #pragma unroll
        for (int t = 0; t < 6; ++t)
            #pragma unroll
            for (int r = 0; r < 8; ++r) O[t][r] *= alpha;

        // Build P B-fragment (32 i x 16 j) from the two C-layout S tiles.
        float q0[8], q1[8];
        #pragma unroll
        for (int r = 0; r < 8; ++r) { q0[r] = __shfl_xor(p0[r], 16, 32); q1[r] = __shfl_xor(p1[r], 16, 32); }
        v8u pu;
        #pragma unroll
        for (int r = 0; r < 4; ++r) {
            float a0 = hi ? q1[2*r]   : p0[2*r];
            float a1 = hi ? q1[2*r+1] : p0[2*r+1];
            float b0 = hi ? p1[2*r]   : q0[2*r];
            float b1 = hi ? p1[2*r+1] : q0[2*r+1];
            pu[r]     = pk_bf16(a0, a1);   // K 0..7  (lo) / K16..23 (hi)
            pu[4 + r] = pk_bf16(b0, b1);   // K 8..15 (lo) / K24..31 (hi)
        }
        v16bf pf = __builtin_bit_cast(v16bf, pu);
        #pragma unroll
        for (int dt = 0; dt < 6; ++dt) {
            v16bf va = load_a(Vb + (size_t)dt * 16 * (M_ * 2) + i0 * 2, M_ * 2, lane);
            O[dt] = wmma_bf16(va, pf, O[dt]);
        }
    }
    float inv = 1.0f / lrun;
    int j = j0 + (lane & 15);
    int dofs = (lane >> 4) * 8;
    unsigned short* arow = aT + ((size_t)b * M_ + j) * C_ + (size_t)h * DH_;
    #pragma unroll
    for (int dt = 0; dt < 6; ++dt)
        #pragma unroll
        for (int r = 0; r < 8; ++r)
            arow[dt * 16 + dofs + r] = f2bf(O[dt][r] * inv);
}

// ---------------- Kernel 5: proj GEMM (WMMA) + residual -> y (fp32) in d_out ----------------
__global__ void k_proj_gemm(const char* __restrict__ wpb, const char* __restrict__ aT,
                            const float* __restrict__ x, float* __restrict__ y) {
    int lane = threadIdx.x & 31;
    int wid  = blockIdx.x * 8 + (threadIdx.x >> 5);
    int ms = wid & 7;
    int ot = (wid >> 3) % 48;
    int b  = wid / (48 * 8);
    int o0 = ot * 16, mbase = ms * 64;

    const char* Abase = wpb + (size_t)o0 * (C_ * 2);
    const char* Bbase = aT + ((size_t)b * M_ + mbase) * (C_ * 2);
    v8f acc[4];
    #pragma unroll
    for (int s = 0; s < 4; ++s) acc[s] = (v8f)(0.0f);

    for (int ck = 0; ck < C_ / 32; ++ck) {
        v16bf af = load_a(Abase + ck * 64, C_ * 2, lane);
        #pragma unroll
        for (int s = 0; s < 4; ++s) {
            v16bf bf = load_b(Bbase + (size_t)s * 16 * (C_ * 2) + ck * 64, C_ * 2, lane);
            acc[s] = wmma_bf16(af, bf, acc[s]);
        }
    }
    int lm = lane & 15, oofs = (lane >> 4) * 8;
    #pragma unroll
    for (int s = 0; s < 4; ++s) {
        int m = mbase + s * 16 + lm;
        #pragma unroll
        for (int r = 0; r < 8; ++r) {
            int o = o0 + oofs + r;
            size_t idx = ((size_t)b * C_ + o) * M_ + m;
            y[idx] = acc[s][r] + x[idx];
        }
    }
}

// ---------------- Kernel 6: layernorm over C (ddof=1) + exp(log_scale)*y + bias, in place ----------------
__global__ void k_layernorm(float* __restrict__ y, const float* __restrict__ log_scale,
                            const float* __restrict__ bias) {
    int tid = blockIdx.x * 256 + threadIdx.x;   // one (b, m) column each
    int b = tid / M_, m = tid % M_;
    size_t base = (size_t)b * C_ * M_ + m;
    float s = 0.0f, s2 = 0.0f;
    for (int c = 0; c < C_; ++c) {
        float v = y[base + (size_t)c * M_];
        s += v; s2 += v * v;
    }
    float mu  = s / (float)C_;
    float var = (s2 - (float)C_ * mu * mu) / (float)(C_ - 1);
    float inv = 1.0f / (sqrtf(var) + EPS_);
    for (int c = 0; c < C_; ++c) {
        size_t idx = base + (size_t)c * M_;
        float v = (y[idx] - mu) * inv;
        y[idx] = __expf(log_scale[c]) * v + bias[c];
    }
}

extern "C" void kernel_launch(void* const* d_in, const int* in_sizes, int n_in,
                              void* d_out, int out_size, void* d_ws, size_t ws_size,
                              hipStream_t stream) {
    const float* x         = (const float*)d_in[0];
    const float* w_qkv     = (const float*)d_in[1];
    const float* w_proj    = (const float*)d_in[2];
    const float* log_scale = (const float*)d_in[3];
    const float* bias      = (const float*)d_in[4];
    float* out = (float*)d_out;

    char* ws = (char*)d_ws;
    // bf16 workspace layout (16B aligned)
    unsigned short* wqb = (unsigned short*)(ws + 0);           //  2304*768
    unsigned short* wpb = (unsigned short*)(ws + 3538944);     //   768*768
    unsigned short* xT  = (unsigned short*)(ws + 4718592);     // B*M*C
    unsigned short* qT  = (unsigned short*)(ws + 55050240);    // B*NH*M*DH
    unsigned short* kT  = (unsigned short*)(ws + 105381888);   // B*NH*M*DH
    unsigned short* vm  = (unsigned short*)(ws + 155713536);   // B*NH*DH*M
    unsigned short* aT  = (unsigned short*)(ws + 206045184);   // B*M*C

    k_cvt_weights<<<1024, 256, 0, stream>>>(w_qkv, w_proj, wqb, wpb);
    k_transpose_x<<<dim3(M_ / 32, C_ / 32, B_), dim3(32, 8), 0, stream>>>(x, xT);
    k_qkv_gemm<<<(64 * 144 * 8) / 8, 256, 0, stream>>>((const char*)wqb, (const char*)xT, qT, kT, vm);
    k_attention<<<(64 * 8 * 32) / 8, 256, 0, stream>>>((const char*)qT, (const char*)kT, (const char*)vm, aT);
    k_proj_gemm<<<(64 * 48 * 8) / 8, 256, 0, stream>>>((const char*)wpb, (const char*)aT, x, out);
    k_layernorm<<<(B_ * M_) / 256, 256, 0, stream>>>(out, log_scale, bias);
}
